// MultiHeadAttention_44272522887234
// MI455X (gfx1250) — matrix operand order repaired and checked by reading, not compiled, not run on hardware
//
#include <hip/hip_runtime.h>
#include <stdint.h>

// ---------------------------------------------------------------------------
// CDNA5 (gfx1250) wave32 WMMA multi-head attention forward.
//   B=2, S=2048, E=2048, H=16, D=128
// bf16 matmuls with f32 accumulation via v_wmma_f32_16x16x32_bf16.
// GEMMs double-buffer LDS tiles with GLOBAL_LOAD_ASYNC_TO_LDS (ASYNCcnt)
// when the toolchain exposes the builtin, else sync copies.
// ---------------------------------------------------------------------------

typedef __attribute__((ext_vector_type(16))) __bf16        v16bf;
typedef __attribute__((ext_vector_type(8)))  float         v8f;
typedef __attribute__((ext_vector_type(8)))  unsigned int  v8u;
typedef __attribute__((ext_vector_type(2)))  int           v2i;

#ifndef __has_builtin
#define __has_builtin(x) 0
#endif
#if __has_builtin(__builtin_amdgcn_global_load_async_to_lds_b64)
#define ASYNC_LDS 1
#else
#define ASYNC_LDS 0
#endif

__device__ __forceinline__ uint16_t f2bf(float f) {
  uint32_t u = __builtin_bit_cast(uint32_t, f);
  return (uint16_t)((u + 0x7FFFu + ((u >> 16) & 1u)) >> 16);  // RNE
}
__device__ __forceinline__ v16bf frag(v8u u) { return __builtin_bit_cast(v16bf, u); }
__device__ __forceinline__ v8f zero8() {
  v8f z;
  #pragma unroll
  for (int i = 0; i < 8; i++) z[i] = 0.0f;
  return z;
}

// 8-byte global -> LDS copy (async when available)
__device__ __forceinline__ void copy8(const uint16_t* g, uint16_t* l) {
#if ASYNC_LDS
  __builtin_amdgcn_global_load_async_to_lds_b64(
      (__attribute__((address_space(1))) v2i*)g,
      (__attribute__((address_space(3))) v2i*)l, 0, 0);
#else
  *(uint2*)l = *(const uint2*)g;
#endif
}
__device__ __forceinline__ void async_wait() {
#if ASYNC_LDS
#if __has_builtin(__builtin_amdgcn_s_wait_asynccnt)
  __builtin_amdgcn_s_wait_asynccnt(0);
#else
  asm volatile("s_wait_asynccnt 0x0" ::: "memory");
#endif
#endif
}

// ---------------------------------------------------------------------------
// elementwise f32 -> bf16
// ---------------------------------------------------------------------------
__global__ void cvt_bf16_kernel(const float* __restrict__ in,
                                uint16_t* __restrict__ out, int n) {
  int i = blockIdx.x * blockDim.x + threadIdx.x;
  int stride = gridDim.x * blockDim.x;
  for (; i < n; i += stride) out[i] = f2bf(in[i]);
}

// ---------------------------------------------------------------------------
// C = A[M,K] @ W[N,K]^T   (A, W bf16 row-major; both read rows along K)
// MODE 0: bf16 output scattered head-major:  out[((b*H+h)*S+s)*D+d]
// MODE 1: f32 output row-major [M,N], plus bias[n]
// Block tile 128(M) x 256(N), BK=32, double-buffered LDS.
// 8 waves as 2(M) x 4(N); wave tile 64x64 = 4x4 WMMA tiles (128 acc VGPRs).
// ---------------------------------------------------------------------------
template <int MODE>
__global__ __launch_bounds__(256)
void gemm_bt_kernel(const uint16_t* __restrict__ A, const uint16_t* __restrict__ W,
                    const float* __restrict__ bias, void* __restrict__ outv,
                    int M, int N, int K, int S, int H, int D) {
  constexpr int LDT = 36;           // u16 LDS row stride (+4 pad, 8B-aligned rows)
  constexpr int LDT32 = LDT / 2;
  __shared__ __align__(16) uint16_t As[2 * 128 * LDT];   // 18.4 KB
  __shared__ __align__(16) uint16_t Bs[2 * 256 * LDT];   // 36.9 KB

  const int tid = threadIdx.x;
  const int wave = tid >> 5, lane = tid & 31, lo = lane & 15, hi = lane >> 4;
  const int wm = wave >> 2;  // 0..1 : 64 rows
  const int wn = wave & 3;   // 0..3 : 64 cols
  const int m0 = blockIdx.y * 128;
  const int n0 = blockIdx.x * 256;

  v8f acc[4][4];
  #pragma unroll
  for (int i = 0; i < 4; i++)
    #pragma unroll
    for (int j = 0; j < 4; j++) acc[i][j] = zero8();

  auto load_tiles = [&](int buf, int kk0) {
    uint16_t* Asb = As + buf * (128 * LDT);
    uint16_t* Bsb = Bs + buf * (256 * LDT);
    #pragma unroll
    for (int c0 = 0; c0 < 4; c0++) {            // A: 128 rows x 32 = 1024 8B chunks
      int c = tid + c0 * 256;
      int row = c >> 3, seg = c & 7;
      copy8(&A[(size_t)(m0 + row) * K + kk0 + seg * 4], &Asb[row * LDT + seg * 4]);
    }
    #pragma unroll
    for (int c0 = 0; c0 < 8; c0++) {            // W: 256 rows x 32 = 2048 8B chunks
      int c = tid + c0 * 256;
      int row = c >> 3, seg = c & 7;
      copy8(&W[(size_t)(n0 + row) * K + kk0 + seg * 4], &Bsb[row * LDT + seg * 4]);
    }
  };

  const int KT = K / 32;
  load_tiles(0, 0);                              // prologue: tile 0

  for (int kt = 0; kt < KT; kt++) {
    async_wait();          // my async loads into buf[kt&1] have landed
    __syncthreads();       // everyone's have; everyone done with buf[1-cur] reads
    const int cur = kt & 1;
    if (kt + 1 < KT) load_tiles(1 - cur, (kt + 1) * 32);   // overlap with WMMA

    const uint32_t* As32 = (const uint32_t*)(As + cur * (128 * LDT));
    const uint32_t* Bs32 = (const uint32_t*)(Bs + cur * (256 * LDT));

    v8u a[4], b[4];
    #pragma unroll
    for (int tm = 0; tm < 4; tm++) {
      int r = wm * 64 + tm * 16 + lo;            // A: row-per-lane, K split by hi
      #pragma unroll
      for (int i = 0; i < 8; i++)
        a[tm][i] = As32[r * LDT32 + ((i < 4) ? 0 : 8) + hi * 4 + (i & 3)];
    }
    #pragma unroll
    for (int tn = 0; tn < 4; tn++) {
      int r = wn * 64 + tn * 16 + lo;            // B: column-per-lane, K split by hi
      #pragma unroll
      for (int i = 0; i < 8; i++)
        b[tn][i] = Bs32[r * LDT32 + ((i < 4) ? 0 : 8) + hi * 4 + (i & 3)];
    }
    #pragma unroll
    for (int tm = 0; tm < 4; tm++)
      #pragma unroll
      for (int tn = 0; tn < 4; tn++)
        acc[tm][tn] = __builtin_amdgcn_wmma_f32_16x16x32_bf16(
            false, frag(a[tm]), false, frag(b[tn]), (short)0, acc[tm][tn],
            false, false);
  }

  // C/D layout: N = lane&15 ; M = 8*(lane>>4) + vgpr
  #pragma unroll
  for (int tm = 0; tm < 4; tm++)
    #pragma unroll
    for (int tn = 0; tn < 4; tn++)
      #pragma unroll
      for (int r = 0; r < 8; r++) {
        int m = m0 + wm * 64 + tm * 16 + hi * 8 + r;
        int n = n0 + wn * 64 + tn * 16 + lo;
        float v = acc[tm][tn][r];
        if (MODE == 0) {
          int b_ = m / S, s_ = m % S, h_ = n / D, d_ = n % D;
          ((uint16_t*)outv)[((size_t)(b_ * H + h_) * S + s_) * D + d_] = f2bf(v);
        } else {
          ((float*)outv)[(size_t)m * N + n] = v + bias[n];
        }
      }
}

// ---------------------------------------------------------------------------
// Causal flash attention. Q/K/V head-major bf16 [B*H, S, D], D=128.
// Block: 128 query rows for one (b,h). kv tiles of 64, online softmax.
// ctx written bf16 as [b, s, h*D+d] so the output GEMM can consume it flat.
// ---------------------------------------------------------------------------
__global__ __launch_bounds__(256)
void attn_kernel(const uint16_t* __restrict__ Q, const uint16_t* __restrict__ Kh,
                 const uint16_t* __restrict__ V, uint16_t* __restrict__ ctx,
                 int S, int H, int Dd) {
  constexpr int D = 128, BQ = 128, BKV = 64;
  constexpr int LQ = 132, LQ32 = 66;   // Qs/Ks u16 stride
  constexpr int LV = 68,  LV32 = 34;   // Vt u16 stride
  constexpr int LS = 66;               // Sc f32 stride; bf16-P overlay stride 132 u16
  __shared__ __align__(16) uint16_t Qs[BQ * LQ];
  __shared__ __align__(16) uint16_t Ks[BKV * LQ];
  __shared__ __align__(16) uint16_t Vt[D * LV];    // transposed: Vt[d][kv]
  __shared__ __align__(16) float    Sc[BQ * LS];   // scores f32; P bf16 overlays it
  __shared__ float rowM[BQ], rowL[BQ], rowF[BQ];
  (void)Dd;

  const uint32_t* Qs32 = (const uint32_t*)Qs;
  const uint32_t* Ks32 = (const uint32_t*)Ks;
  const uint32_t* Vt32 = (const uint32_t*)Vt;
  const uint32_t* P32  = (const uint32_t*)Sc;
  uint16_t* P16 = (uint16_t*)Sc;

  const int tid = threadIdx.x;
  const int wave = tid >> 5, lane = tid & 31, lo = lane & 15, hi = lane >> 4;
  const int s_wm = wave >> 1, s_wn = wave & 1;   // S phase: 4(M) x 2(N) waves
  const int p_wm = wave >> 2, p_wn = wave & 3;   // PV phase: 2(M) x 4(N) waves

  const int q0 = blockIdx.x * BQ;
  const int bh = blockIdx.y;
  const int b = bh / H, h = bh % H;
  const uint16_t* Qg = Q  + (size_t)bh * S * D;
  const uint16_t* Kg = Kh + (size_t)bh * S * D;
  const uint16_t* Vg = V  + (size_t)bh * S * D;

  // Q tile: 128x128 bf16 = 4096 8-byte chunks
  for (int c = tid; c < 4096; c += 256) {
    int row = c >> 5, seg = c & 31;
    *(uint2*)&Qs[row * LQ + seg * 4] =
        *(const uint2*)&Qg[(size_t)(q0 + row) * D + seg * 4];
  }
  if (tid < BQ) { rowM[tid] = -1e30f; rowL[tid] = 0.0f; }

  v8f ctxa[4][2];
  #pragma unroll
  for (int i = 0; i < 4; i++)
    #pragma unroll
    for (int j = 0; j < 2; j++) ctxa[i][j] = zero8();

  const float scale = 0.08838834764831845f;  // 1/sqrt(128)
  const int ktMax = q0 / BKV + 1;            // causal: last kv tile touching diag

  for (int kt = 0; kt <= ktMax; kt++) {
    const int k0 = kt * BKV;
    __syncthreads();   // previous iter's P/Vt consumers done
    // K tile 64x128 + V tile transposed into Vt[d][kv]
    for (int c = tid; c < 2048; c += 256) {
      int row = c >> 5, seg = c & 31;
      *(uint2*)&Ks[row * LQ + seg * 4] =
          *(const uint2*)&Kg[(size_t)(k0 + row) * D + seg * 4];
      uint2 v = *(const uint2*)&Vg[(size_t)(k0 + row) * D + seg * 4];
      Vt[(seg * 4 + 0) * LV + row] = (uint16_t)(v.x & 0xffffu);
      Vt[(seg * 4 + 1) * LV + row] = (uint16_t)(v.x >> 16);
      Vt[(seg * 4 + 2) * LV + row] = (uint16_t)(v.y & 0xffffu);
      Vt[(seg * 4 + 3) * LV + row] = (uint16_t)(v.y >> 16);
    }
    __syncthreads();

    // S = Q @ K^T : per wave 2x2 tiles, K-dim 128 = 4 WMMA steps
    v8f sacc[2][2];
    #pragma unroll
    for (int i = 0; i < 2; i++)
      #pragma unroll
      for (int j = 0; j < 2; j++) sacc[i][j] = zero8();
    #pragma unroll
    for (int ks = 0; ks < 4; ks++) {
      v8u a[2], bb[2];
      #pragma unroll
      for (int tm = 0; tm < 2; tm++) {
        int r = s_wm * 32 + tm * 16 + lo;
        #pragma unroll
        for (int i = 0; i < 8; i++)
          a[tm][i] = Qs32[r * LQ32 + ks * 16 + ((i < 4) ? 0 : 8) + hi * 4 + (i & 3)];
      }
      #pragma unroll
      for (int tn = 0; tn < 2; tn++) {
        int r = s_wn * 32 + tn * 16 + lo;
        #pragma unroll
        for (int i = 0; i < 8; i++)
          bb[tn][i] = Ks32[r * LQ32 + ks * 16 + ((i < 4) ? 0 : 8) + hi * 4 + (i & 3)];
      }
      #pragma unroll
      for (int tm = 0; tm < 2; tm++)
        #pragma unroll
        for (int tn = 0; tn < 2; tn++)
          sacc[tm][tn] = __builtin_amdgcn_wmma_f32_16x16x32_bf16(
              false, frag(a[tm]), false, frag(bb[tn]), (short)0, sacc[tm][tn],
              false, false);
    }
    #pragma unroll
    for (int tm = 0; tm < 2; tm++)
      #pragma unroll
      for (int tn = 0; tn < 2; tn++)
        #pragma unroll
        for (int r = 0; r < 8; r++)
          Sc[(s_wm * 32 + tm * 16 + hi * 8 + r) * LS +
             s_wn * 32 + tn * 16 + lo] = sacc[tm][tn][r];
    __syncthreads();

    // online softmax: one thread per q row (in-place bf16 overlay is safe:
    // word j is read before any word <= (j-1)/2 has been overwritten)
    if (tid < BQ) {
      int q = tid, gq = q0 + q;
      int valid = gq - k0 + 1;
      if (valid > BKV) valid = BKV;
      if (valid < 0)   valid = 0;
      float mprev = rowM[q], mnew = mprev;
      for (int j = 0; j < valid; j++)
        mnew = fmaxf(mnew, Sc[q * LS + j] * scale);
      float f = __expf(mprev - mnew);
      float sum = 0.0f;
      for (int j = 0; j < valid; j++) {
        float p = __expf(Sc[q * LS + j] * scale - mnew);
        sum += p;
        P16[q * 2 * LS + j] = f2bf(p);
      }
      for (int j = valid; j < BKV; j++) P16[q * 2 * LS + j] = 0;
      rowM[q] = mnew;
      rowL[q] = rowL[q] * f + sum;
      rowF[q] = f;
    }
    __syncthreads();

    // rescale ctx by exp(m_old - m_new), then ctx += P @ V
    #pragma unroll
    for (int tm = 0; tm < 4; tm++)
      #pragma unroll
      for (int r = 0; r < 8; r++) {
        float f = rowF[p_wm * 64 + tm * 16 + hi * 8 + r];
        #pragma unroll
        for (int tn = 0; tn < 2; tn++) ctxa[tm][tn][r] *= f;
      }
    #pragma unroll
    for (int kk = 0; kk < 2; kk++) {
      v8u a[4], bb[2];
      #pragma unroll
      for (int tm = 0; tm < 4; tm++) {
        int r = p_wm * 64 + tm * 16 + lo;
        #pragma unroll
        for (int i = 0; i < 8; i++)
          a[tm][i] = P32[r * LS + kk * 16 + ((i < 4) ? 0 : 8) + hi * 4 + (i & 3)];
      }
      #pragma unroll
      for (int tn = 0; tn < 2; tn++) {
        int r = p_wn * 32 + tn * 16 + lo;   // column d of B = row d of Vt
        #pragma unroll
        for (int i = 0; i < 8; i++)
          bb[tn][i] = Vt32[r * LV32 + kk * 16 + ((i < 4) ? 0 : 8) + hi * 4 + (i & 3)];
      }
      #pragma unroll
      for (int tm = 0; tm < 4; tm++)
        #pragma unroll
        for (int tn = 0; tn < 2; tn++)
          ctxa[tm][tn] = __builtin_amdgcn_wmma_f32_16x16x32_bf16(
              false, frag(a[tm]), false, frag(bb[tn]), (short)0, ctxa[tm][tn],
              false, false);
    }
  }

  __syncthreads();
  if (tid < BQ) rowF[tid] = 1.0f / rowL[tid];
  __syncthreads();

  // write ctx bf16 at [b, gq, h*D + d]
  #pragma unroll
  for (int tm = 0; tm < 4; tm++)
    #pragma unroll
    for (int r = 0; r < 8; r++) {
      int m = p_wm * 64 + tm * 16 + hi * 8 + r;
      float inv = rowF[m];
      int gq = q0 + m;
      #pragma unroll
      for (int tn = 0; tn < 2; tn++) {
        int d = p_wn * 32 + tn * 16 + lo;
        ctx[((size_t)b * S + gq) * (size_t)(H * D) + h * D + d] =
            f2bf(ctxa[tm][tn][r] * inv);
      }
    }
}

// ---------------------------------------------------------------------------
extern "C" void kernel_launch(void* const* d_in, const int* in_sizes, int n_in,
                              void* d_out, int out_size, void* d_ws, size_t ws_size,
                              hipStream_t stream) {
  (void)in_sizes; (void)n_in; (void)out_size; (void)ws_size;
  const float* x  = (const float*)d_in[0];
  const float* Wq = (const float*)d_in[1];
  const float* Wk = (const float*)d_in[2];
  const float* Wv = (const float*)d_in[3];
  const float* Wo = (const float*)d_in[4];
  const float* bo = (const float*)d_in[5];

  const int B = 2, S = 2048, E = 2048, H = 16, D = 128;
  const int nX = B * S * E;   // 8,388,608
  const int nW = E * E;       // 4,194,304

  char* ws = (char*)d_ws;
  uint16_t* xb   = (uint16_t*)(ws);                 // 16 MB
  uint16_t* ctxb = xb;                              // alias: x dead after QKV
  uint16_t* wqb  = (uint16_t*)(ws + 16777216);      // 8 MB each
  uint16_t* wkb  = (uint16_t*)(ws + 25165824);
  uint16_t* wvb  = (uint16_t*)(ws + 33554432);
  uint16_t* wob  = (uint16_t*)(ws + 41943040);
  uint16_t* qh   = (uint16_t*)(ws + 50331648);      // 16 MB each
  uint16_t* kh   = (uint16_t*)(ws + 67108864);
  uint16_t* vh   = (uint16_t*)(ws + 83886080);      // end: 100,663,296 B

  cvt_bf16_kernel<<<1024, 256, 0, stream>>>(x,  xb,  nX);
  cvt_bf16_kernel<<<512, 256, 0, stream>>>(Wq, wqb, nW);
  cvt_bf16_kernel<<<512, 256, 0, stream>>>(Wk, wkb, nW);
  cvt_bf16_kernel<<<512, 256, 0, stream>>>(Wv, wvb, nW);
  cvt_bf16_kernel<<<512, 256, 0, stream>>>(Wo, wob, nW);

  dim3 g(E / 256, (B * S) / 128);
  gemm_bt_kernel<0><<<g, 256, 0, stream>>>(xb, wqb, nullptr, qh, B * S, E, E, S, H, D);
  gemm_bt_kernel<0><<<g, 256, 0, stream>>>(xb, wkb, nullptr, kh, B * S, E, E, S, H, D);
  gemm_bt_kernel<0><<<g, 256, 0, stream>>>(xb, wvb, nullptr, vh, B * S, E, E, S, H, D);

  attn_kernel<<<dim3(S / 128, B * H), 256, 0, stream>>>(qh, kh, vh, ctxb, S, H, D);

  gemm_bt_kernel<1><<<g, 256, 0, stream>>>(ctxb, wob, bo, d_out, B * S, E, E, S, H, D);
}
